// MultiPhraseAttention_68023692034396
// MI455X (gfx1250) — matrix operand order repaired and checked by reading, not compiled, not run on hardware
//
#include <hip/hip_runtime.h>
#include <hip/hip_bf16.h>
#include <math.h>

typedef __bf16 bf16_t;
typedef __attribute__((ext_vector_type(8)))  bf16_t v8bf;
typedef __attribute__((ext_vector_type(16))) bf16_t v16bf;
typedef __attribute__((ext_vector_type(8)))  float  v8f;

#define SEQ 2048
#define BSZ 2
#define DM  1024
#define NH  16
#define HDD 64
#define PN  256
#define TOK (SEQ*BSZ)

#define WMMA_BF16(a,b,c) \
  __builtin_amdgcn_wmma_f32_16x16x32_bf16(false,(a),false,(b),(short)0,(c),false,false)

static __device__ __forceinline__ v16bf ld16(const bf16_t* p0, const bf16_t* p1) {
  union { v16bf v; v8bf h[2]; } u;
  u.h[0] = *(const v8bf*)p0;
  u.h[1] = *(const v8bf*)p1;
  return u.v;
}
static __device__ __forceinline__ v8f zero8() {
  v8f z;
#pragma unroll
  for (int i = 0; i < 8; i++) z[i] = 0.f;
  return z;
}
static __device__ __forceinline__ float redmax16(float v) {
#pragma unroll
  for (int m = 1; m < 16; m <<= 1) v = fmaxf(v, __shfl_xor(v, m, 32));
  return v;
}
static __device__ __forceinline__ float redsum16(float v) {
#pragma unroll
  for (int m = 1; m < 16; m <<= 1) v += __shfl_xor(v, m, 32);
  return v;
}

// LDS aperture (ISA 10.2): flat address low 32 bits are the LDS byte address.
static __device__ __forceinline__ unsigned lds_off(const void* p) {
  return (unsigned)(unsigned long long)p;
}
// CDNA5 async global->LDS copy, 16B per lane, tracked by ASYNCcnt (ISA 10 / 08_async_tensor).
static __device__ __forceinline__ void async_g2lds_b128(unsigned dst_lds, const void* src) {
  asm volatile("global_load_async_to_lds_b128 %0, %1, off"
               :: "v"(dst_lds), "v"((unsigned long long)src) : "memory");
}
static __device__ __forceinline__ void wait_asynccnt0() {
  asm volatile("s_wait_asynccnt 0" ::: "memory");
}

// ---------------- conversion / packing ----------------
__global__ void mpa_cvt_bf16(const float* __restrict__ s, bf16_t* __restrict__ d, int n) {
  int i = blockIdx.x * 256 + threadIdx.x;
  if (i < n) d[i] = (bf16_t)s[i];
}

// phrase max-pool over window of 8: Pbf[(p*B+b)*D + d]
__global__ void mpa_phrase(const float* __restrict__ q, bf16_t* __restrict__ P) {
  int i = blockIdx.x * 256 + threadIdx.x;
  if (i >= PN * BSZ * DM) return;
  int dcol = i & (DM - 1);
  int row  = i >> 10;           // p*BSZ + b
  int b = row & (BSZ - 1);
  int p = row >> 1;
  const float* base = q + ((size_t)(p * 8) * BSZ + b) * DM + dcol;
  float m = base[0];
#pragma unroll
  for (int w = 1; w < 8; ++w) m = fmaxf(m, base[(size_t)w * BSZ * DM]);
  P[i] = (bf16_t)m;
}

// ---------------- WMMA GEMM with async-LDS double-buffered W panel ----------------
// C = (A @ W^T + bias) * scale.  Block: 64M x 64N, 4 waves (one 16x64 each).
// W tile (64 rows x 32 k = 4KB) staged in LDS via GLOBAL_LOAD_ASYNC_TO_LDS_B128,
// double buffered; A streamed to registers with 1-iteration prefetch.
// All 4 B tiles are fetched to registers before the WMMA group so the DS pipe
// overlaps the matrix pipe (partial s_wait_dscnt instead of 4x full drains).
// mode 0: bf16 out, head layout ((b*NH+h)*Slen + s)*HDD + hd   (s = m/BSZ, b = m%BSZ)
// mode 1: bf16 out, transposed head layout ((b*NH+h)*HDD + hd)*Slen + s
// mode 2: fp32 out, flat m*N + n
__global__ void mpa_gemm_wmma(const bf16_t* __restrict__ A, const bf16_t* __restrict__ W,
                              const float* __restrict__ bias, void* __restrict__ out,
                              int M, int N, int K, int mode, float scale, int Slen) {
  __shared__ __align__(16) bf16_t btile[2][64][32];

  int wave = threadIdx.x >> 5, lane = threadIdx.x & 31;
  int l15 = lane & 15, hf = lane >> 4;
  int tilesN = N >> 6;
  int bm = blockIdx.x / tilesN, bn = blockIdx.x % tilesN;
  int n0 = bn * 64;
  int mrow0 = bm * 64 + wave * 16;                 // this wave's 16 M-rows
  const bf16_t* arow = A + (size_t)(mrow0 + l15) * K;

  // async-copy lane mapping: 2 instructions per wave cover the 64x32 tile
  int flat0 = (wave * 2) * 32 + lane;              // 16B quanta index
  int row0 = flat0 >> 2, kq0 = flat0 & 3;
  int flat1 = flat0 + 32;
  int row1 = flat1 >> 2, kq1 = flat1 & 3;

  auto issue_b = [&](int kk, int buf) {
    async_g2lds_b128(lds_off(&btile[buf][row0][kq0 * 8]),
                     W + (size_t)(n0 + row0) * K + kk + kq0 * 8);
    async_g2lds_b128(lds_off(&btile[buf][row1][kq1 * 8]),
                     W + (size_t)(n0 + row1) * K + kk + kq1 * 8);
  };

  v8f c[4];
#pragma unroll
  for (int j = 0; j < 4; j++) c[j] = zero8();

  issue_b(0, 0);
  v16bf aCur = ld16(arow + hf * 8, arow + 16 + hf * 8);

  int cur = 0;
  for (int kk = 0; kk < K; kk += 32) {
    wait_asynccnt0();          // my async loads for 'cur' complete
    __syncthreads();           // everyone's loads visible; prev buffer fully consumed
    int nxt = cur ^ 1;
    v16bf aNext = aCur;
    if (kk + 32 < K) {
      issue_b(kk + 32, nxt);
      aNext = ld16(arow + kk + 32 + hf * 8, arow + kk + 48 + hf * 8);
    }
    // batch-issue all B-tile DS loads, then run the WMMA group
    v16bf bt[4];
#pragma unroll
    for (int j = 0; j < 4; j++) {
      const bf16_t* bp = &btile[cur][j * 16 + l15][hf * 8];
      bt[j] = ld16(bp, bp + 16);
    }
#pragma unroll
    for (int j = 0; j < 4; j++) c[j] = WMMA_BF16(aCur, bt[j], c[j]);
    aCur = aNext;
    cur = nxt;
  }

#pragma unroll
  for (int j = 0; j < 4; j++) {
    int n = n0 + j * 16 + l15;
    float bn = bias[n];
#pragma unroll
    for (int r = 0; r < 8; r++) {
      float val = (c[j][r] + bn) * scale;
      int m = mrow0 + r + hf * 8;
      if (mode == 2) {
        ((float*)out)[(size_t)m * N + n] = val;
      } else {
        int h = n >> 6, hd = n & 63;
        int s = m >> 1, bb = m & 1;   // BSZ == 2
        size_t addr = (mode == 0)
            ? ((size_t)(bb * NH + h) * Slen + s) * HDD + hd
            : ((size_t)(bb * NH + h) * HDD + hd) * (size_t)Slen + s;
        ((bf16_t*)out)[addr] = (bf16_t)val;
      }
    }
  }
}

// ---------------- Gv = gauss @ v  (sigma=2 -> +/-16 window) ----------------
// VT: (B*NH, HDD, SEQ) bf16.  GvT: (B*NH, HDD, PN) bf16.
__global__ void mpa_gv(const bf16_t* __restrict__ VT, bf16_t* __restrict__ GvT) {
  int hd = threadIdx.x;                 // 64
  int p  = blockIdx.x & (PN - 1);
  int bh = blockIdx.x >> 8;
  const bf16_t* vrow = VT + ((size_t)bh * HDD + hd) * SEQ;
  float mu = (float)(p * 8) + 3.5f;
  float acc = 0.f;
  int j0 = p * 8 - 12, j1 = p * 8 + 19; // exp(-30) tails negligible
  for (int j = j0; j <= j1; ++j) {
    if (j < 0 || j >= SEQ) continue;
    float d = (float)j - mu;
    acc += __expf(-0.125f * d * d) * 0.19947114020071635f * (float)vrow[j];
  }
  GvT[((size_t)bh * HDD + hd) * PN + p] = (bf16_t)acc;
}

// ---------------- flash branch: o = softmax(q K^T) @ V (V given transposed) ----------------
static __device__ __forceinline__ void mpa_flash(
    const bf16_t* __restrict__ Keys,   // (len x HDD) row-major
    const bf16_t* __restrict__ ValsT,  // (HDD x len) row-major
    int len, v16bf qa0, v16bf qa1,
    bf16_t (*pl)[32], int l15, int hf, v8f o[4]) {
  float m8[8], l8[8];
#pragma unroll
  for (int r = 0; r < 8; r++) { m8[r] = -3.0e38f; l8[r] = 0.f; }
#pragma unroll
  for (int j = 0; j < 4; j++) o[j] = zero8();

  for (int kb = 0; kb < len; kb += 32) {
    const bf16_t* kp0 = Keys + (size_t)(kb + l15) * HDD;
    const bf16_t* kp1 = Keys + (size_t)(kb + 16 + l15) * HDD;
    // batch the 4 key B-tile loads before the score WMMA group
    v16bf kb00 = ld16(kp0 + hf * 8,      kp0 + 16 + hf * 8);
    v16bf kb01 = ld16(kp0 + 32 + hf * 8, kp0 + 48 + hf * 8);
    v16bf kb10 = ld16(kp1 + hf * 8,      kp1 + 16 + hf * 8);
    v16bf kb11 = ld16(kp1 + 32 + hf * 8, kp1 + 48 + hf * 8);
    v8f s0 = zero8(), s1 = zero8();
    s0 = WMMA_BF16(qa0, kb00, s0);
    s0 = WMMA_BF16(qa1, kb01, s0);
    s1 = WMMA_BF16(qa0, kb10, s1);
    s1 = WMMA_BF16(qa1, kb11, s1);

    float p0[8], p1[8];
#pragma unroll
    for (int r = 0; r < 8; r++) {
      float bm = redmax16(fmaxf(s0[r], s1[r]));
      float mn = fmaxf(m8[r], bm);
      float corr = __expf(m8[r] - mn);
      m8[r] = mn;
      p0[r] = __expf(s0[r] - mn);
      p1[r] = __expf(s1[r] - mn);
      float rs = redsum16(p0[r] + p1[r]);
      l8[r] = l8[r] * corr + rs;
#pragma unroll
      for (int j = 0; j < 4; j++) o[j][r] *= corr;
    }
    // stage P (C layout -> row-major LDS -> A layout)
#pragma unroll
    for (int r = 0; r < 8; r++) {
      int mrow = r + hf * 8;
      pl[mrow][l15]      = (bf16_t)p0[r];
      pl[mrow][16 + l15] = (bf16_t)p1[r];
    }
    asm volatile("s_wait_dscnt 0" ::: "memory");
    v16bf pa = ld16(&pl[l15][hf * 16], &pl[l15][hf * 16 + 8]);
    // batch the 4 V B-tile loads before the PV WMMA group
    v16bf vt[4];
#pragma unroll
    for (int j = 0; j < 4; j++) {
      const bf16_t* vp = ValsT + (size_t)(j * 16 + l15) * len + kb + hf * 16;
      vt[j] = ld16(vp, vp + 8);
    }
#pragma unroll
    for (int j = 0; j < 4; j++) o[j] = WMMA_BF16(pa, vt[j], o[j]);
  }
#pragma unroll
  for (int r = 0; r < 8; r++) {
    float inv = 1.f / l8[r];
#pragma unroll
    for (int j = 0; j < 4; j++) o[j][r] *= inv;
  }
}

// ---------------- fused dual-branch attention ----------------
__global__ void mpa_attn(const bf16_t* __restrict__ Qb, const bf16_t* __restrict__ Kb,
                         const bf16_t* __restrict__ Qg, const bf16_t* __restrict__ Kg,
                         const bf16_t* __restrict__ VT, const bf16_t* __restrict__ GvT,
                         bf16_t* __restrict__ Obuf) {
  __shared__ __align__(16) bf16_t pl[4][16][32];
  int wave = threadIdx.x >> 5, lane = threadIdx.x & 31;
  int l15 = lane & 15, hf = lane >> 4;
  int g = blockIdx.x * 4 + wave;
  int qt = g & (SEQ / 16 - 1);
  int bh = g / (SEQ / 16);
  int s0 = qt * 16;

  const bf16_t* qbp = Qb + ((size_t)bh * SEQ + s0 + l15) * HDD;
  v16bf qb0 = ld16(qbp + hf * 8,      qbp + 16 + hf * 8);
  v16bf qb1 = ld16(qbp + 32 + hf * 8, qbp + 48 + hf * 8);
  const bf16_t* qgp = Qg + ((size_t)bh * SEQ + s0 + l15) * HDD;
  v16bf qg0 = ld16(qgp + hf * 8,      qgp + 16 + hf * 8);
  v16bf qg1 = ld16(qgp + 32 + hf * 8, qgp + 48 + hf * 8);

  v8f og[4], ob[4];
  mpa_flash(Kg + (size_t)bh * PN * HDD,  GvT + (size_t)bh * HDD * PN,  PN,  qg0, qg1, pl[wave], l15, hf, og);
  mpa_flash(Kb + (size_t)bh * SEQ * HDD, VT  + (size_t)bh * HDD * SEQ, SEQ, qb0, qb1, pl[wave], l15, hf, ob);

  int b = bh >> 4, h = bh & 15;
#pragma unroll
  for (int j = 0; j < 4; j++)
#pragma unroll
    for (int r = 0; r < 8; r++) {
      int m = r + hf * 8;
      int t = (s0 + m) * BSZ + b;
      int col = h * HDD + j * 16 + l15;
      Obuf[(size_t)t * DM + col] = (bf16_t)(0.5f * (ob[j][r] + og[j][r]));
    }
}

// ---------------- host launcher ----------------
extern "C" void kernel_launch(void* const* d_in, const int* in_sizes, int n_in,
                              void* d_out, int out_size, void* d_ws, size_t ws_size,
                              hipStream_t stream) {
  (void)in_sizes; (void)n_in; (void)out_size; (void)ws_size;
  const float* query   = (const float*)d_in[0];
  const float* Wq_base = (const float*)d_in[1];
  const float* bq_base = (const float*)d_in[2];
  const float* Wk_base = (const float*)d_in[3];
  const float* bk_base = (const float*)d_in[4];
  const float* Wq_g    = (const float*)d_in[5];
  const float* bq_g    = (const float*)d_in[6];
  const float* Wk_g    = (const float*)d_in[7];
  const float* bk_g    = (const float*)d_in[8];
  const float* Wv      = (const float*)d_in[9];
  const float* bv      = (const float*)d_in[10];
  const float* Wo      = (const float*)d_in[11];
  const float* bo      = (const float*)d_in[12];

  char* ws = (char*)d_ws;
  size_t off = 0;
  auto alloc = [&](size_t bytes) -> void* {
    void* p = ws + off;
    off += (bytes + 255) & ~(size_t)255;
    return p;
  };
  bf16_t* Xbf  = (bf16_t*)alloc((size_t)TOK * DM * 2);
  bf16_t* Pbf  = (bf16_t*)alloc((size_t)PN * BSZ * DM * 2);
  bf16_t* Wqb  = (bf16_t*)alloc((size_t)DM * DM * 2);
  bf16_t* Wkb  = (bf16_t*)alloc((size_t)DM * DM * 2);
  bf16_t* Wqg  = (bf16_t*)alloc((size_t)DM * DM * 2);
  bf16_t* Wkg  = (bf16_t*)alloc((size_t)DM * DM * 2);
  bf16_t* Wvb  = (bf16_t*)alloc((size_t)DM * DM * 2);
  bf16_t* Wob  = (bf16_t*)alloc((size_t)DM * DM * 2);
  bf16_t* Qb   = (bf16_t*)alloc((size_t)BSZ * NH * SEQ * HDD * 2);
  bf16_t* Kb   = (bf16_t*)alloc((size_t)BSZ * NH * SEQ * HDD * 2);
  bf16_t* Qg   = (bf16_t*)alloc((size_t)BSZ * NH * SEQ * HDD * 2);
  bf16_t* Vt   = (bf16_t*)alloc((size_t)BSZ * NH * SEQ * HDD * 2);
  bf16_t* Kg   = (bf16_t*)alloc((size_t)BSZ * NH * PN * HDD * 2);
  bf16_t* Gvt  = (bf16_t*)alloc((size_t)BSZ * NH * PN * HDD * 2);
  bf16_t* Obuf = (bf16_t*)alloc((size_t)TOK * DM * 2);

  int n = TOK * DM;
  mpa_cvt_bf16<<<(n + 255) / 256, 256, 0, stream>>>(query, Xbf, n);
  mpa_phrase<<<(PN * BSZ * DM + 255) / 256, 256, 0, stream>>>(query, Pbf);
  n = DM * DM;
  mpa_cvt_bf16<<<(n + 255) / 256, 256, 0, stream>>>(Wq_base, Wqb, n);
  mpa_cvt_bf16<<<(n + 255) / 256, 256, 0, stream>>>(Wk_base, Wkb, n);
  mpa_cvt_bf16<<<(n + 255) / 256, 256, 0, stream>>>(Wq_g,    Wqg, n);
  mpa_cvt_bf16<<<(n + 255) / 256, 256, 0, stream>>>(Wk_g,    Wkg, n);
  mpa_cvt_bf16<<<(n + 255) / 256, 256, 0, stream>>>(Wv,      Wvb, n);
  mpa_cvt_bf16<<<(n + 255) / 256, 256, 0, stream>>>(Wo,      Wob, n);

  const float kScale = 0.125f;  // HD^-0.5, folded into Q projections
  // blocks = (M/64)*(N/64): 4096 rows -> 1024 blocks; 512 rows -> 128 blocks
  mpa_gemm_wmma<<<1024, 128, 0, stream>>>(Xbf, Wqb, bq_base, Qb, TOK, DM, DM, 0, kScale, SEQ);
  mpa_gemm_wmma<<<1024, 128, 0, stream>>>(Xbf, Wkb, bk_base, Kb, TOK, DM, DM, 0, 1.0f,   SEQ);
  mpa_gemm_wmma<<<1024, 128, 0, stream>>>(Xbf, Wqg, bq_g,    Qg, TOK, DM, DM, 0, kScale, SEQ);
  mpa_gemm_wmma<<<1024, 128, 0, stream>>>(Xbf, Wvb, bv,      Vt, TOK, DM, DM, 1, 1.0f,   SEQ);
  mpa_gemm_wmma<<<128,  128, 0, stream>>>(Pbf, Wkg, bk_g,    Kg, PN * BSZ, DM, DM, 0, 1.0f, PN);

  mpa_gv<<<BSZ * NH * PN, 64, 0, stream>>>(Vt, Gvt);

  // B*H*(S/16)/4 = 1024 blocks
  mpa_attn<<<1024, 128, 0, stream>>>(Qb, Kb, Qg, Kg, Vt, Gvt, Obuf);

  mpa_gemm_wmma<<<1024, 128, 0, stream>>>(Obuf, Wob, bo, d_out, TOK, DM, DM, 2, 1.0f, SEQ);
}
